// LightweightEncoder_42013370090023
// MI455X (gfx1250) — matrix operand order repaired and checked by reading, not compiled, not run on hardware
//
#include <hip/hip_runtime.h>
#include <math.h>

typedef __attribute__((ext_vector_type(16))) _Float16 v16h;
typedef __attribute__((ext_vector_type(8)))  _Float16 v8h;
typedef __attribute__((ext_vector_type(8)))  float    v8f;
typedef __attribute__((ext_vector_type(4)))  float    f4;
typedef __attribute__((ext_vector_type(4)))  unsigned int u32x4;
typedef __attribute__((ext_vector_type(8)))  int      i32x8;
typedef __attribute__((ext_vector_type(4)))  int      i32x4;

#ifndef __has_builtin
#define __has_builtin(x) 0
#endif
#if __has_builtin(__builtin_amdgcn_tensor_load_to_lds) && __has_builtin(__builtin_amdgcn_s_wait_tensorcnt)
#define USE_TDM 1
#else
#define USE_TDM 0
#endif

// ---------------- workspace layout (in 4-byte units) ----------------
static const size_t OFF_SIGMA = 0;                       // 1
static const size_t OFF_WN    = 16;                      // 144  (W/sigma, [3][48])
static const size_t OFF_WPAD  = 160;                     // 512 floats = 16x64 f16 padded A
static const size_t OFF_X     = 768;                     // 8*3*256*256
static const size_t OFF_H0    = OFF_X   + 1572864ULL;    // 8*3*512*512
static const size_t OFF_H1B   = OFF_H0  + 6291456ULL;    // 8*3*256*256
static const size_t OFF_H2    = OFF_H1B + 1572864ULL;    // 8*3*128*128 (h)
static const size_t OFF_H3    = OFF_H2  + 393216ULL;     // 8*3*64*64   (h1)
static const size_t OFF_F1    = OFF_H3  + 98304ULL;      // feat_1 [8,3,128,128]
static const size_t OFF_F2    = OFF_F1  + 393216ULL;     // feat_2 [8,3,64,64]
static const size_t OFF_P1    = OFF_F2  + 98304ULL;      // p1 [8,3,128,128]
static const size_t OFF_P2    = OFF_P1  + 393216ULL;     // p2 [8,3,64,64]
static const size_t OFF_ACC   = OFF_P2  + 98304ULL;      // 11 uints: esti0..2, hist[8]

// ---------------- init ----------------
__global__ void zero_acc_kernel(unsigned int* acc) {
    if (threadIdx.x < 11) acc[threadIdx.x] = 0u;
}

// ---- sigma (largest singular value of 3x48) + normalized & f16-padded weights ----
__global__ void prep_sigma_kernel(const float* __restrict__ sw,
                                  float* __restrict__ wn,
                                  _Float16* __restrict__ wpad,   // [16][64], zero padded
                                  float* __restrict__ sigma_out) {
    __shared__ float s_inv;
    if (threadIdx.x == 0) {
        float G[3][3];
        for (int i = 0; i < 3; ++i)
            for (int j = 0; j < 3; ++j) {
                float s = 0.f;
                for (int k = 0; k < 48; ++k) s += sw[i * 48 + k] * sw[j * 48 + k];
                G[i][j] = s;
            }
        float v0 = 1.f, v1 = 1.f, v2 = 1.f, lam = 0.f;
        for (int it = 0; it < 64; ++it) {
            float u0 = G[0][0] * v0 + G[0][1] * v1 + G[0][2] * v2;
            float u1 = G[1][0] * v0 + G[1][1] * v1 + G[1][2] * v2;
            float u2 = G[2][0] * v0 + G[2][1] * v1 + G[2][2] * v2;
            lam = sqrtf(u0 * u0 + u1 * u1 + u2 * u2);
            float inv = 1.0f / fmaxf(lam, 1e-30f);
            v0 = u0 * inv; v1 = u1 * inv; v2 = u2 * inv;
        }
        float sigma = sqrtf(fmaxf(lam, 1e-30f));
        sigma_out[0] = sigma;
        s_inv = 1.0f / sigma;
    }
    __syncthreads();
    float inv = s_inv;
    for (int i = threadIdx.x; i < 144; i += blockDim.x) wn[i] = sw[i] * inv;
    for (int i = threadIdx.x; i < 16 * 64; i += blockDim.x) {
        int row = i >> 6, K = i & 63;
        float v = (row < 3 && K < 48) ? sw[row * 48 + K] * inv : 0.0f;
        wpad[i] = (_Float16)v;
    }
}

// ---------------- sample conv via WMMA, input band staged through LDS by the TDM.
// Block = one output row (b, oy): stages 3ch x 4rows x 1024cols (48 KB) into LDS with
// one TENSOR_LOAD_TO_LDS, then 8 waves x 2 tiles x (2 WMMA) consume it from LDS.
__global__ void sample_conv_wmma_kernel(const float* __restrict__ xin,
                                        const _Float16* __restrict__ wpad, // [16][64]
                                        const float* __restrict__ bias,
                                        float* __restrict__ xout) {
    __shared__ float tile[3 * 4 * 1024];   // [ch][row][col]

    const int lane  = threadIdx.x & 31;
    const int wave  = threadIdx.x >> 5;
    const int col   = lane & 15;           // N column == A row
    const int khalf = (lane >> 4) & 1;     // which K half this lane holds

    const int b  = blockIdx.x >> 8;        // 8 batches
    const int oy = blockIdx.x & 255;       // 256 output rows

    const float* gp0 = xin + (size_t)b * 3145728ULL + (size_t)oy * 4096ULL;

#if USE_TDM
    if (threadIdx.x < 32) {
        unsigned long long ga = (unsigned long long)(const char*)gp0;   // byte address
        unsigned ldsoff = (unsigned)(unsigned long long)(void*)&tile[0];
        u32x4 g0;
        g0[0] = 1u;                                   // count=1, user mode
        g0[1] = ldsoff;                               // lds_addr (bytes)
        g0[2] = (unsigned)(ga & 0xFFFFFFFFu);         // global_addr[31:0]
        g0[3] = ((unsigned)(ga >> 32) & 0x01FFFFFFu)  // global_addr[56:32]
                | (2u << 30);                         // type = 2 (image)
        i32x8 g1;
        g1[0] = (int)(2u << 16);          // wg_mask=0, data_size=2 (4 bytes)
        g1[1] = (int)(1024u << 16);       // tensor_dim0[15:0] = 1024
        g1[2] = (int)(1024u << 16);       // tensor_dim0 hi=0 | tensor_dim1[15:0]=1024
        g1[3] = (int)(1024u << 16);       // tensor_dim1 hi=0 | tile_dim0=1024
        g1[4] = (int)(4u | (3u << 16));   // tile_dim1=4, tile_dim2=3
        g1[5] = 1024;                     // tensor_dim0_stride lo32 = 1024
        g1[6] = 0;                        // dim0_stride hi | dim1_stride[15:0]=0
        g1[7] = 16;                       // tensor_dim1_stride[47:16] = 1048576>>16
        i32x4 g2;
        g2[0] = 3;                        // tensor_dim2 = 3
        g2[1] = 0;                        // tensor_dim3 = 0
        g2[2] = 0x100000;                 // tensor_dim2_stride = 1048576
        g2[3] = 0;                        // tile_dim3 = 0
        i32x4 g3; g3[0] = 0; g3[1] = 0; g3[2] = 0; g3[3] = 0;
#if __clang_major__ >= 23
        i32x8 g4;
        g4[0]=0; g4[1]=0; g4[2]=0; g4[3]=0; g4[4]=0; g4[5]=0; g4[6]=0; g4[7]=0;
        __builtin_amdgcn_tensor_load_to_lds(g0, g1, g2, g3, g4, 0);
#else
        __builtin_amdgcn_tensor_load_to_lds(g0, g1, g2, g3, 0);
#endif
        __builtin_amdgcn_s_wait_tensorcnt(0);
    }
    __syncthreads();
#else
    // fallback: cooperative float4 copy global -> LDS
    for (int i = threadIdx.x; i < 3072; i += 256) {
        int row = i >> 8, j = i & 255;        // 12 rows of 256 float4
        int ch = row >> 2, r = row & 3;
        f4 v = *(const f4*)(gp0 + (size_t)ch * 1048576ULL + (size_t)r * 1024 + j * 4);
        ((f4*)tile)[i] = v;
    }
    __syncthreads();
#endif

    // ---- A fragments from padded f16 weights: 4 unconditional 16B loads ----
    const v8h* wp = (const v8h*)(wpad + col * 64);
    v8h A00 = wp[2 * khalf];
    v8h A01 = wp[2 * khalf + 1];
    v8h A10 = wp[4 + 2 * khalf];
    v8h A11 = wp[5 + 2 * khalf];
    v16h a0 = __builtin_shufflevector(A00, A01, 0,1,2,3,4,5,6,7,8,9,10,11,12,13,14,15);
    v16h a1 = __builtin_shufflevector(A10, A11, 0,1,2,3,4,5,6,7,8,9,10,11,12,13,14,15);

    const float b0c = bias[0], b1c = bias[1], b2c = bias[2];

#pragma unroll
    for (int t = 0; t < 2; ++t) {
        int ox = (wave * 2 + t) * 16 + col;    // 8 waves x 2 tiles cover 256 pixels

        // ---- B fragment 0: channel khalf, 4x4 patch from LDS ----
        const float* tc = tile + khalf * 4096 + ox * 4;
        f4 r0 = *(const f4*)(tc);
        f4 r1 = *(const f4*)(tc + 1024);
        f4 r2 = *(const f4*)(tc + 2048);
        f4 r3 = *(const f4*)(tc + 3072);
        v16h bf0;
#pragma unroll
        for (int e = 0; e < 4; ++e) {
            bf0[e]      = (_Float16)r0[e];
            bf0[4 + e]  = (_Float16)r1[e];
            bf0[8 + e]  = (_Float16)r2[e];
            bf0[12 + e] = (_Float16)r3[e];
        }

        // ---- B fragment 1: K+32 -> channel 2 (valid only for khalf==0) ----
        v16h bf1;
        if (khalf == 0) {
            const float* t2 = tile + 2 * 4096 + ox * 4;
            f4 s0 = *(const f4*)(t2);
            f4 s1 = *(const f4*)(t2 + 1024);
            f4 s2 = *(const f4*)(t2 + 2048);
            f4 s3 = *(const f4*)(t2 + 3072);
#pragma unroll
            for (int e = 0; e < 4; ++e) {
                bf1[e]      = (_Float16)s0[e];
                bf1[4 + e]  = (_Float16)s1[e];
                bf1[8 + e]  = (_Float16)s2[e];
                bf1[12 + e] = (_Float16)s3[e];
            }
        } else {
#pragma unroll
            for (int e = 0; e < 16; ++e) bf1[e] = (_Float16)0.0f;
        }

        v8f acc = {};
        acc = __builtin_amdgcn_wmma_f32_16x16x32_f16(false, a0, false, bf0, (short)0, acc, false, false);
        acc = __builtin_amdgcn_wmma_f32_16x16x32_f16(false, a1, false, bf1, (short)0, acc, false, false);

        if (lane < 16) {   // lanes 0..15 hold D rows 0..7 -> channels 0..2 in acc[0..2]
            size_t ob = (((size_t)b * 3) * 256 + oy) * 256 + ox;
            xout[ob]          = acc[0] + b0c;
            xout[ob + 65536]  = acc[1] + b1c;
            xout[ob + 131072] = acc[2] + b2c;
        }
    }
}

// ---------------- ctx layer: conv3x3 stride2 pad1 of relu(bn(in)); optional tanh feature
__global__ void ctx_conv_kernel(const float* __restrict__ in, float* __restrict__ out,
                                float* __restrict__ feat,
                                const float* __restrict__ w,   // [3][3][3][3]
                                const float* __restrict__ cb,  // [3]
                                const float* __restrict__ g,   // [3]
                                const float* __restrict__ bb,  // [3]
                                int Hi, int Wi, int Ho, int Wo) {
    int idx = blockIdx.x * blockDim.x + threadIdx.x;
    int total = 8 * 3 * Ho * Wo;
    if (idx >= total) return;
    int ox = idx % Wo; int t = idx / Wo;
    int oy = t % Ho;   t /= Ho;
    int oc = t % 3;    int b = t / 3;

    const float rinv = rsqrtf(1.0f + 0.001f);
    float sum = cb[oc];
#pragma unroll
    for (int ic = 0; ic < 3; ++ic) {
        float inv = g[ic] * rinv, beta = bb[ic];
        const float* ip = in + ((size_t)(b * 3 + ic)) * Hi * Wi;
        const float* wp = w + (oc * 3 + ic) * 9;
#pragma unroll
        for (int ky = 0; ky < 3; ++ky) {
            int iy = oy * 2 - 1 + ky;
            if (iy < 0 || iy >= Hi) continue;
#pragma unroll
            for (int kx = 0; kx < 3; ++kx) {
                int ix = ox * 2 - 1 + kx;
                if (ix < 0 || ix >= Wi) continue;
                float v = ip[(size_t)iy * Wi + ix];
                float a = fmaxf(inv * v + beta, 0.0f);
                sum += a * wp[ky * 3 + kx];
            }
        }
    }
    out[idx] = sum;
    if (feat) feat[idx] = (tanhf(sum) + 1.0f) * 0.5f;
}

// ---------------- pool conv: 2x2 stride2 pad0, 3->3 channels
__global__ void pool_conv_kernel(const float* __restrict__ in, float* __restrict__ out,
                                 const float* __restrict__ w,  // [3][3][2][2]
                                 const float* __restrict__ cb,
                                 int Hi, int Wi, int Ho, int Wo) {
    int idx = blockIdx.x * blockDim.x + threadIdx.x;
    int total = 8 * 3 * Ho * Wo;
    if (idx >= total) return;
    int ox = idx % Wo; int t = idx / Wo;
    int oy = t % Ho;   t /= Ho;
    int oc = t % 3;    int b = t / 3;

    float sum = cb[oc];
#pragma unroll
    for (int ic = 0; ic < 3; ++ic) {
        const float* ip = in + ((size_t)(b * 3 + ic)) * Hi * Wi;
        const float* wp = w + (oc * 3 + ic) * 4;
#pragma unroll
        for (int ky = 0; ky < 2; ++ky)
#pragma unroll
            for (int kx = 0; kx < 2; ++kx)
                sum += ip[(size_t)(oy * 2 + ky) * Wi + (ox * 2 + kx)] * wp[ky * 2 + kx];
    }
    out[idx] = sum;
}

// ---------------- fused mode-select + soft VQ + reductions at 256-res
__global__ void quant_main_kernel(const float* __restrict__ x,
                                  const float* __restrict__ f1, const float* __restrict__ f2,
                                  const float* __restrict__ p1, const float* __restrict__ p2,
                                  const float* __restrict__ thresh,
                                  const float* __restrict__ centers,
                                  float* __restrict__ soft_out, unsigned int* __restrict__ acc) {
    __shared__ unsigned int s_acc[11];
    if (threadIdx.x < 11) s_acc[threadIdx.x] = 0u;
    __syncthreads();

    int idx = blockIdx.x * blockDim.x + threadIdx.x;
    const int total = 8 * 3 * 256 * 256;
    if (idx < total) {
        float th1 = thresh[0], th2 = thresh[1];
        int xx = idx & 255; int t = idx >> 8;
        int yy = t & 255;   t >>= 8;           // t = b*3 + c
        size_t base64  = (size_t)t * 64 * 64;
        size_t base128 = (size_t)t * 128 * 128;
        float f2v = f2[base64  + (size_t)(yy >> 2) * 64  + (xx >> 2)];
        float f1v = f1[base128 + (size_t)(yy >> 1) * 128 + (xx >> 1)];
        bool cond2 = f2v < th2;
        bool cond1 = (!cond2) && (f1v < th1);
        bool cond0 = !(cond1 || cond2);
        float xv = x[idx];
        float xq = cond2 ? p2[base64  + (size_t)(yy >> 2) * 64  + (xx >> 2)]
                 : cond1 ? p1[base128 + (size_t)(yy >> 1) * 128 + (xx >> 1)]
                 : xv;
        float d[8]; float dmin = 1e30f; int kmin = 0;
#pragma unroll
        for (int k = 0; k < 8; ++k) {
            float dd = xq - centers[k]; d[k] = dd * dd;
            if (d[k] < dmin) { dmin = d[k]; kmin = k; }
        }
        float wsum = 0.f, ssum = 0.f;
#pragma unroll
        for (int k = 0; k < 8; ++k) {
            float e = __expf(-(d[k] - dmin));
            wsum += e; ssum += centers[k] * e;
        }
        soft_out[idx] = ssum / wsum;
        if (cond0) { atomicAdd(&s_acc[0], 1u); atomicAdd(&s_acc[3 + kmin], 1u); }
        else if (cond1) atomicAdd(&s_acc[1], 1u);
        else            atomicAdd(&s_acc[2], 1u);
    }
    __syncthreads();
    if (threadIdx.x < 11 && s_acc[threadIdx.x]) atomicAdd(&acc[threadIdx.x], s_acc[threadIdx.x]);
}

// ---------------- masked quant-index histograms for p1 (mode 1) / p2 (mode 2)
__global__ void quant_hist_kernel(const float* __restrict__ t,
                                  const float* __restrict__ f1, const float* __restrict__ f2,
                                  const float* __restrict__ thresh,
                                  const float* __restrict__ centers,
                                  unsigned int* __restrict__ acc, int mode, int HW) {
    __shared__ unsigned int s_h[8];
    if (threadIdx.x < 8) s_h[threadIdx.x] = 0u;
    __syncthreads();

    int idx = blockIdx.x * blockDim.x + threadIdx.x;
    int total = 8 * 3 * HW * HW;
    if (idx < total) {
        float th1 = thresh[0], th2 = thresh[1];
        int xx = idx % HW; int tt = idx / HW;
        int yy = tt % HW;  tt /= HW;           // tt = b*3 + c
        bool mask;
        if (mode == 1) {  // 128-res: up2(feat_2) >= th2 && feat_1 < th1
            float f2v = f2[(size_t)tt * 64 * 64 + (size_t)(yy >> 1) * 64 + (xx >> 1)];
            mask = (f2v >= th2) && (f1[idx] < th1);
        } else {          // 64-res: feat_2 < th2
            mask = f2[idx] < th2;
        }
        if (mask) {
            float v = t[idx]; float dmin = 1e30f; int kmin = 0;
#pragma unroll
            for (int k = 0; k < 8; ++k) {
                float dd = v - centers[k]; dd *= dd;
                if (dd < dmin) { dmin = dd; kmin = k; }
            }
            atomicAdd(&s_h[kmin], 1u);
        }
    }
    __syncthreads();
    if (threadIdx.x < 8 && s_h[threadIdx.x]) atomicAdd(&acc[3 + threadIdx.x], s_h[threadIdx.x]);
}

// ---------------- finalize scalars
__global__ void finalize_kernel(const unsigned int* __restrict__ acc, float* __restrict__ out2) {
    if (threadIdx.x == 0 && blockIdx.x == 0) {
        float e0 = (float)acc[0], e1 = (float)acc[1], e2 = (float)acc[2];
        float esti_size = e0 + e1 * 0.25f + e2 * 0.0625f;
        float esti_cr = (1.0f / 16.0f) * esti_size / (256.0f * 256.0f * 3.0f * 8.0f);
        float tot = 0.f, cnt[8];
        for (int k = 0; k < 8; ++k) { cnt[k] = (float)acc[3 + k]; tot += cnt[k]; }
        float var = 0.f;
        if (tot > 0.f) {
            const float m = 0.125f;  // normalized counts sum to 1 -> mean = 1/8
            for (int k = 0; k < 8; ++k) { float c = cnt[k] / tot - m; var += c * c; }
            var /= 7.0f;             // ddof = 1
        }
        out2[0] = esti_cr;
        out2[1] = sqrtf(var);
    }
}

static inline unsigned int nblk(int n, int b) { return (unsigned int)((n + b - 1) / b); }

extern "C" void kernel_launch(void* const* d_in, const int* in_sizes, int n_in,
                              void* d_out, int out_size, void* d_ws, size_t ws_size,
                              hipStream_t stream) {
    (void)in_sizes; (void)n_in; (void)out_size; (void)ws_size;

    const float* x_init   = (const float*)d_in[0];
    const float* thresh   = (const float*)d_in[1];
    const float* sample_w = (const float*)d_in[2];
    const float* sample_b = (const float*)d_in[3];
    const float* centers  = (const float*)d_in[4];
    const float* pool1_w  = (const float*)d_in[5];
    const float* pool1_b  = (const float*)d_in[6];
    const float* pool2_w  = (const float*)d_in[7];
    const float* pool2_b  = (const float*)d_in[8];
    const float* ctx_w    = (const float*)d_in[9];   // [4][3][3][3][3]
    const float* ctx_b    = (const float*)d_in[10];  // [4][3]
    const float* ctx_g    = (const float*)d_in[11];  // [4][3]
    const float* ctx_bb   = (const float*)d_in[12];  // [4][3]
    float* out = (float*)d_out;

    float* ws = (float*)d_ws;
    _Float16* wpad    = (_Float16*)(ws + OFF_WPAD);
    unsigned int* acc = (unsigned int*)(ws + OFF_ACC);

    zero_acc_kernel<<<1, 32, 0, stream>>>(acc);
    prep_sigma_kernel<<<1, 256, 0, stream>>>(sample_w, ws + OFF_WN, wpad, ws + OFF_SIGMA);

    // sample conv (TDM -> LDS -> WMMA): one block per (b, oy) output row
    sample_conv_wmma_kernel<<<2048, 256, 0, stream>>>(x_init, wpad, sample_b, ws + OFF_X);

    // context extractor (4 layers); layers 2/3 also emit tanh feature maps
    ctx_conv_kernel<<<nblk(8*3*512*512, 256), 256, 0, stream>>>(
        x_init, ws + OFF_H0, nullptr, ctx_w + 0,   ctx_b + 0, ctx_g + 0, ctx_bb + 0, 1024, 1024, 512, 512);
    ctx_conv_kernel<<<nblk(8*3*256*256, 256), 256, 0, stream>>>(
        ws + OFF_H0, ws + OFF_H1B, nullptr, ctx_w + 81,  ctx_b + 3, ctx_g + 3, ctx_bb + 3, 512, 512, 256, 256);
    ctx_conv_kernel<<<nblk(8*3*128*128, 256), 256, 0, stream>>>(
        ws + OFF_H1B, ws + OFF_H2, ws + OFF_F1, ctx_w + 162, ctx_b + 6, ctx_g + 6, ctx_bb + 6, 256, 256, 128, 128);
    ctx_conv_kernel<<<nblk(8*3*64*64, 256), 256, 0, stream>>>(
        ws + OFF_H2, ws + OFF_H3, ws + OFF_F2, ctx_w + 243, ctx_b + 9, ctx_g + 9, ctx_bb + 9, 128, 128, 64, 64);

    // pooling pyramid
    pool_conv_kernel<<<nblk(8*3*128*128, 256), 256, 0, stream>>>(
        ws + OFF_X,  ws + OFF_P1, pool1_w, pool1_b, 256, 256, 128, 128);
    pool_conv_kernel<<<nblk(8*3*64*64, 256), 256, 0, stream>>>(
        ws + OFF_P1, ws + OFF_P2, pool2_w, pool2_b, 128, 128, 64, 64);

    // fused quantization + reductions
    quant_main_kernel<<<nblk(8*3*256*256, 256), 256, 0, stream>>>(
        ws + OFF_X, ws + OFF_F1, ws + OFF_F2, ws + OFF_P1, ws + OFF_P2,
        thresh, centers, out, acc);
    quant_hist_kernel<<<nblk(8*3*128*128, 256), 256, 0, stream>>>(
        ws + OFF_P1, ws + OFF_F1, ws + OFF_F2, thresh, centers, acc, 1, 128);
    quant_hist_kernel<<<nblk(8*3*64*64, 256), 256, 0, stream>>>(
        ws + OFF_P2, ws + OFF_F1, ws + OFF_F2, thresh, centers, acc, 2, 64);

    finalize_kernel<<<1, 32, 0, stream>>>(acc, out + 1572864);
}
